// DepthWiseSeparableAttention_51161650430466
// MI455X (gfx1250) — matrix operand order repaired and checked by reading, not compiled, not run on hardware
//
#include <hip/hip_runtime.h>
#include <hip/hip_bf16.h>
#include <math.h>

// ---------------------------------------------------------------------------
// DepthWiseSeparableAttention for MI455X (gfx1250, wave32, WMMA bf16)
//
// Pipeline (bias path eliminated: softmax(a + rowconst) == softmax(a)):
//   1. ln_kernel        : LayerNorm(x) -> xn (bf16)
//   2. prep_qkv_w       : fold BN + attn-scale into w_qkv -> bf16 B-matrix
//   3. prep_proj_w      : w_proj -> bf16 B-matrix
//   4. qkv_gemm         : xn @ w_qkv^T (+folded bias) -> q[h][n][d], k[h][n][d],
//                         vT[h][d][n]   (all bf16, v_wmma_f32_16x16x32_bf16)
//   5. attn_kernel      : per (head, 16-query tile): full 4096-key score row
//                         resident in 256KB LDS -> softmax -> PV (wmma),
//                         cross-wave reduce via ds_add_f32
//   6. proj_kernel      : out = x + ao @ w_proj^T + b_proj   (wmma)
// ---------------------------------------------------------------------------

typedef __attribute__((ext_vector_type(16))) __bf16 v16bf;
typedef __attribute__((ext_vector_type(8)))  __bf16 v8bf;
typedef __attribute__((ext_vector_type(8)))  float  v8f;

#define DIMC     256
#define HEADS    8
#define HEAD_DIM 32
#define N_TOK    4096
#define QKV_DIM  768
#define ATTN_SCALE 0.17677669529663687f  /* 32^-0.5 */

static __device__ __forceinline__ v8f wmma_bf16(v16bf a, v16bf b, v8f c) {
  // D = A(16x32) * B(32x16) + C(16x16 f32)
  return __builtin_amdgcn_wmma_f32_16x16x32_bf16(false, a, false, b, (short)0, c,
                                                 false, false);
}

// A-fragment (16x32 bf16) per-lane layout (contiguous K, matches B):
//   lanes 0-15 : M=lane,   K = {0..15}
//   lanes16-31 : M=lane-16,K = {16..31}
static __device__ __forceinline__ v16bf load_a_frag_bf16(const __bf16* row, int half) {
  v8bf a0 = *(const v8bf*)(row + half * 16);
  v8bf a1 = *(const v8bf*)(row + half * 16 + 8);
  v16bf a;
#pragma unroll
  for (int i = 0; i < 8; i++) { a[i] = a0[i]; a[i + 8] = a1[i]; }
  return a;
}

// ---------------------------------------------------------------------------
// 1. LayerNorm: one block (256 threads) per token row
// ---------------------------------------------------------------------------
__global__ void ln_kernel(const float* __restrict__ x, const float* __restrict__ g,
                          const float* __restrict__ b, __bf16* __restrict__ xn) {
  __shared__ float red[DIMC];
  const int row = blockIdx.x, tid = threadIdx.x;
  const float v = x[(size_t)row * DIMC + tid];
  red[tid] = v;
  __syncthreads();
  for (int s = 128; s > 0; s >>= 1) { if (tid < s) red[tid] += red[tid + s]; __syncthreads(); }
  const float mu = red[0] * (1.0f / DIMC);
  __syncthreads();
  const float d = v - mu;
  red[tid] = d * d;
  __syncthreads();
  for (int s = 128; s > 0; s >>= 1) { if (tid < s) red[tid] += red[tid + s]; __syncthreads(); }
  const float var = red[0] * (1.0f / DIMC);
  const float r = rsqrtf(var + 1e-6f);
  xn[(size_t)row * DIMC + tid] = (__bf16)(d * r * g[tid] + b[tid]);
}

// ---------------------------------------------------------------------------
// 2. Fold BatchNorm (eval) + attention scale into qkv weights/bias, cast bf16
// ---------------------------------------------------------------------------
__global__ void prep_qkv_w(const float* __restrict__ w, const float* __restrict__ bq,
                           const float* __restrict__ bng, const float* __restrict__ bnb,
                           const float* __restrict__ bnm, const float* __restrict__ bnv,
                           __bf16* __restrict__ wb, float* __restrict__ bias) {
  const int i = blockIdx.x * 256 + threadIdx.x;  // over 768*256
  const int d = i >> 8, c = i & 255;
  const float sc = bng[d] * rsqrtf(bnv[d] + 1e-5f);
  const float qs = (d < DIMC) ? ATTN_SCALE : 1.0f;  // fold d^-0.5 into Q rows
  wb[i] = (__bf16)(w[i] * sc * qs);
  if (c == 0) bias[d] = ((bq[d] - bnm[d]) * sc + bnb[d]) * qs;
}

__global__ void prep_proj_w(const float* __restrict__ w, __bf16* __restrict__ wb) {
  const int i = blockIdx.x * 256 + threadIdx.x;  // over 256*256
  wb[i] = (__bf16)w[i];
}

// ---------------------------------------------------------------------------
// 4. qkv GEMM: [4096 x 768] = xn[4096 x 256] @ W^T, scatter into q/k/vT (bf16)
//    grid (256, 12), block 128 (4 waves); each wave owns one 16x16 tile.
// ---------------------------------------------------------------------------
__global__ __launch_bounds__(128) void qkv_gemm(
    const __bf16* __restrict__ xn, const __bf16* __restrict__ wb,
    const float* __restrict__ bias, __bf16* __restrict__ q,
    __bf16* __restrict__ k, __bf16* __restrict__ vT) {
  const int lane = threadIdx.x & 31, wid = threadIdx.x >> 5;
  const int m0 = blockIdx.x * 16;
  const int n0 = (blockIdx.y * 4 + wid) * 16;  // output-channel tile base
  const int col = lane & 15, half = lane >> 4;

  const __bf16* arow = xn + (size_t)(m0 + col) * DIMC;
  const __bf16* brow = wb + (size_t)(n0 + col) * DIMC;  // B col = out-channel row of W
  v8f acc = {};
#pragma unroll
  for (int k0 = 0; k0 < DIMC; k0 += 32) {
    v16bf a = load_a_frag_bf16(arow + k0, half);
    v16bf b = *(const v16bf*)(brow + k0 + half * 16);
    acc = wmma_bf16(a, b, acc);
  }
  const int d = n0 + col;           // this lane's (fixed) output channel
  const float bs = bias[d];
  const int t = d >> 8, hh = (d >> 5) & 7, e = d & 31;
#pragma unroll
  for (int r = 0; r < 8; r++) {
    const int m = m0 + r + half * 8;  // C layout: M = r + 8*(lane/16)
    const __bf16 bv = (__bf16)(acc[r] + bs);
    if (t == 0)      q [((size_t)hh * N_TOK + m) * HEAD_DIM + e] = bv;
    else if (t == 1) k [((size_t)hh * N_TOK + m) * HEAD_DIM + e] = bv;
    else             vT[((size_t)hh * HEAD_DIM + e) * N_TOK + m] = bv;  // transposed
  }
}

// ---------------------------------------------------------------------------
// 5. Attention: block = (head h, 16 queries), 4 waves.
//    LDS: 16 x 4096 fp32 scores (256 KB) + 16x32 fp32 out accumulator.
// ---------------------------------------------------------------------------
__global__ __launch_bounds__(128) void attn_kernel(
    const __bf16* __restrict__ q, const __bf16* __restrict__ k,
    const __bf16* __restrict__ vT, __bf16* __restrict__ ao) {
  extern __shared__ float sc[];                 // [16][4096] + [512]
  float* outacc = sc + 16 * N_TOK;
  const int lane = threadIdx.x & 31, wid = threadIdx.x >> 5;
  const int h = blockIdx.y, m0 = blockIdx.x * 16;
  const int col = lane & 15, half = lane >> 4;

  for (int i = threadIdx.x; i < 512; i += 128) outacc[i] = 0.0f;

  // Q fragment: 16 queries x full head_dim (K=32) in one frag; shared by waves.
  const __bf16* qrow = q + ((size_t)h * N_TOK + m0 + col) * HEAD_DIM;
  const v16bf aq = load_a_frag_bf16(qrow, half);

  // ---- S = Q K^T : each wave covers key chunks nb % 4 == wid ----
  const __bf16* kbase = k + (size_t)h * N_TOK * HEAD_DIM;
  for (int nb = wid; nb < N_TOK / 16; nb += 4) {
    const int key0 = nb * 16;
    const int pf = (key0 + 64 < N_TOK) ? key0 + 64 : key0;
    __builtin_prefetch(kbase + (size_t)(pf + col) * HEAD_DIM, 0, 0);  // global_prefetch_b8
    const v16bf bk = *(const v16bf*)(kbase + (size_t)(key0 + col) * HEAD_DIM + half * 16);
    v8f c = {};
    c = wmma_bf16(aq, bk, c);
#pragma unroll
    for (int r = 0; r < 8; r++)
      sc[(size_t)(r + half * 8) * N_TOK + key0 + col] = c[r];
  }
  __syncthreads();

  // ---- softmax over the full resident 4096-key row (rows 4*wid..4*wid+3) ----
  for (int rr = 0; rr < 4; rr++) {
    float* srow = sc + (size_t)(wid * 4 + rr) * N_TOK;
    float mx = -3.4e38f;
    for (int j = lane; j < N_TOK; j += 32) mx = fmaxf(mx, srow[j]);
    for (int off = 16; off > 0; off >>= 1) mx = fmaxf(mx, __shfl_xor(mx, off, 32));
    float sum = 0.0f;
    for (int j = lane; j < N_TOK; j += 32) {
      const float e = __expf(srow[j] - mx);
      srow[j] = e;
      sum += e;
    }
    for (int off = 16; off > 0; off >>= 1) sum += __shfl_xor(sum, off, 32);
    const float rinv = 1.0f / sum;
    for (int j = lane; j < N_TOK; j += 32) srow[j] *= rinv;
  }
  __syncthreads();

  // ---- out(16x32) = P(16x4096) @ V(4096x32); wave owns keys [wid*1024, +1024) ----
  v8f c0 = {}, c1 = {};
  const __bf16* vbase = vT + (size_t)h * HEAD_DIM * N_TOK;
  for (int kk = wid * 1024; kk < wid * 1024 + 1024; kk += 32) {
    // A = P chunk from LDS (fp32 -> bf16), same split-K lane layout
    const float* prow = sc + (size_t)col * N_TOK + kk + half * 16;
    const float4 p0 = *(const float4*)(prow);
    const float4 p1 = *(const float4*)(prow + 4);
    const float4 p2 = *(const float4*)(prow + 8);
    const float4 p3 = *(const float4*)(prow + 12);
    v16bf ap;
    ap[0]  = (__bf16)p0.x; ap[1]  = (__bf16)p0.y; ap[2]  = (__bf16)p0.z; ap[3]  = (__bf16)p0.w;
    ap[4]  = (__bf16)p1.x; ap[5]  = (__bf16)p1.y; ap[6]  = (__bf16)p1.z; ap[7]  = (__bf16)p1.w;
    ap[8]  = (__bf16)p2.x; ap[9]  = (__bf16)p2.y; ap[10] = (__bf16)p2.z; ap[11] = (__bf16)p2.w;
    ap[12] = (__bf16)p3.x; ap[13] = (__bf16)p3.y; ap[14] = (__bf16)p3.z; ap[15] = (__bf16)p3.w;
    // B = V columns: vT[dim][key] -> contiguous along keys
    const v16bf bv0 = *(const v16bf*)(vbase + (size_t)(col)      * N_TOK + kk + half * 16);
    const v16bf bv1 = *(const v16bf*)(vbase + (size_t)(16 + col) * N_TOK + kk + half * 16);
    c0 = wmma_bf16(ap, bv0, c0);
    c1 = wmma_bf16(ap, bv1, c1);
  }
  // cross-wave reduction in LDS (ds_add_f32)
#pragma unroll
  for (int r = 0; r < 8; r++) {
    const int m = r + half * 8;
    atomicAdd(&outacc[m * 32 + col], c0[r]);
    atomicAdd(&outacc[m * 32 + 16 + col], c1[r]);
  }
  __syncthreads();
  for (int i = threadIdx.x; i < 512; i += 128) {
    const int m = i >> 5, e = i & 31;
    ao[(size_t)(m0 + m) * DIMC + h * HEAD_DIM + e] = (__bf16)outacc[i];
  }
}

// ---------------------------------------------------------------------------
// 6. Projection + residual: out = x + ao @ w_proj^T + b_proj
//    grid (256, 4), block 128; wave owns one 16x16 tile.
// ---------------------------------------------------------------------------
__global__ __launch_bounds__(128) void proj_kernel(
    const __bf16* __restrict__ ao, const __bf16* __restrict__ wp,
    const float* __restrict__ bp, const float* __restrict__ x,
    float* __restrict__ out) {
  const int lane = threadIdx.x & 31, wid = threadIdx.x >> 5;
  const int m0 = blockIdx.x * 16;
  const int n0 = (blockIdx.y * 4 + wid) * 16;
  const int col = lane & 15, half = lane >> 4;

  const __bf16* arow = ao + (size_t)(m0 + col) * DIMC;
  const __bf16* brow = wp + (size_t)(n0 + col) * DIMC;
  v8f acc = {};
#pragma unroll
  for (int k0 = 0; k0 < DIMC; k0 += 32) {
    v16bf a = load_a_frag_bf16(arow + k0, half);
    v16bf b = *(const v16bf*)(brow + k0 + half * 16);
    acc = wmma_bf16(a, b, acc);
  }
  const int cout = n0 + col;
  const float bias = bp[cout];
#pragma unroll
  for (int r = 0; r < 8; r++) {
    const int m = m0 + r + half * 8;
    out[(size_t)m * DIMC + cout] = x[(size_t)m * DIMC + cout] + acc[r] + bias;
  }
}

// ---------------------------------------------------------------------------
extern "C" void kernel_launch(void* const* d_in, const int* in_sizes, int n_in,
                              void* d_out, int out_size, void* d_ws, size_t ws_size,
                              hipStream_t stream) {
  const float* x      = (const float*)d_in[0];
  const float* ln_g   = (const float*)d_in[1];
  const float* ln_b   = (const float*)d_in[2];
  const float* w_qkv  = (const float*)d_in[3];
  const float* b_qkv  = (const float*)d_in[4];
  const float* bn_g   = (const float*)d_in[5];
  const float* bn_b   = (const float*)d_in[6];
  const float* bn_m   = (const float*)d_in[7];
  const float* bn_v   = (const float*)d_in[8];
  // d_in[9] = w_local: provably dead (softmax row-constant invariance)
  const float* w_proj = (const float*)d_in[10];
  const float* b_proj = (const float*)d_in[11];
  float* out = (float*)d_out;

  char* ws = (char*)d_ws;
  size_t off = 0;
  __bf16* xn      = (__bf16*)(ws + off); off += (size_t)N_TOK * DIMC * 2;      // 2 MB
  __bf16* wqkv_b  = (__bf16*)(ws + off); off += (size_t)QKV_DIM * DIMC * 2;    // 384 KB
  float*  qkv_bias= (float*) (ws + off); off += (size_t)QKV_DIM * 4;           // 3 KB
  off = (off + 255) & ~(size_t)255;
  __bf16* wproj_b = (__bf16*)(ws + off); off += (size_t)DIMC * DIMC * 2;       // 128 KB
  __bf16* qb      = (__bf16*)(ws + off); off += (size_t)HEADS * N_TOK * HEAD_DIM * 2;
  __bf16* kb      = (__bf16*)(ws + off); off += (size_t)HEADS * N_TOK * HEAD_DIM * 2;
  __bf16* vTb     = (__bf16*)(ws + off); off += (size_t)HEADS * N_TOK * HEAD_DIM * 2;
  __bf16* ao      = (__bf16*)(ws + off); off += (size_t)N_TOK * DIMC * 2;
  (void)ws_size; (void)in_sizes; (void)n_in; (void)out_size;

  ln_kernel<<<N_TOK, DIMC, 0, stream>>>(x, ln_g, ln_b, xn);
  prep_qkv_w<<<(QKV_DIM * DIMC) / 256, 256, 0, stream>>>(w_qkv, b_qkv, bn_g, bn_b,
                                                         bn_m, bn_v, wqkv_b, qkv_bias);
  prep_proj_w<<<(DIMC * DIMC) / 256, 256, 0, stream>>>(w_proj, wproj_b);
  qkv_gemm<<<dim3(N_TOK / 16, QKV_DIM / 64), 128, 0, stream>>>(xn, wqkv_b, qkv_bias,
                                                               qb, kb, vTb);
  const size_t attn_lds = (size_t)(16 * N_TOK + 512) * sizeof(float);  // 258 KB < 320 KB
  attn_kernel<<<dim3(N_TOK / 16, HEADS), 128, attn_lds, stream>>>(qb, kb, vTb, ao);
  proj_kernel<<<dim3(N_TOK / 16, DIMC / 64), 128, 0, stream>>>(ao, wproj_b, b_proj, x, out);
}
